// CSRSparsity_71562745086422
// MI455X (gfx1250) — matrix operand order repaired and checked by reading, not compiled, not run on hardware
//
#include <hip/hip_runtime.h>
#include <hip/hip_bf16.h>

// ---------------------------------------------------------------------------
// Problem constants (match the reference)
// ---------------------------------------------------------------------------
static constexpr int kB    = 8192;   // batch
static constexpr int kD    = 1024;   // input dim
static constexpr int kH    = 4096;   // hidden dim
static constexpr int kTopK = 8;
static constexpr int kAux  = 512;
static constexpr int kDead = 30;

typedef __attribute__((ext_vector_type(16))) __bf16          v16bf;
typedef __attribute__((ext_vector_type(16))) unsigned short  v16us;
typedef __attribute__((ext_vector_type(8)))  float           v8f;

// ---------------------------------------------------------------------------
// bf16 helpers (round-to-nearest-even split: f = hi + lo, each bf16)
// ---------------------------------------------------------------------------
__device__ __forceinline__ unsigned short f2bf_rne(float f) {
  unsigned int u = __float_as_uint(f);
  unsigned int r = u + 0x7fffu + ((u >> 16) & 1u);
  return (unsigned short)(r >> 16);
}
__device__ __forceinline__ float bf2f(unsigned short h) {
  return __uint_as_float(((unsigned int)h) << 16);
}

// A-fragment K layout within each 32-wide K group: natural [k0..31] so a
// lane's 16-element fragment is one contiguous 32B chunk matching B:
// half==0 -> k[0..15], half==1 -> k[16..31].
__device__ __forceinline__ int a_swz(int kk) {
  return kk;
}

// ---------------------------------------------------------------------------
// Prep kernel A: xs = x - pre_bias, split to bf16 hi/lo, natural [row][k] layout.
// One-shot (≈50MB traffic) — removes all conversion VALU from the GEMM loop,
// which was otherwise repeated once per N-block (64x redundant).
// ---------------------------------------------------------------------------
__global__ void convert_x(const float* __restrict__ x,
                          const float* __restrict__ pre_bias,
                          unsigned short* __restrict__ xhi,
                          unsigned short* __restrict__ xlo)
{
  size_t i = (size_t)blockIdx.x * blockDim.x + threadIdx.x;
  size_t stride = (size_t)gridDim.x * blockDim.x;
  const size_t n = (size_t)kB * kD;
  for (; i < n; i += stride) {
    int k = (int)(i & (size_t)(kD - 1));
    float a = x[i] - pre_bias[k];
    unsigned short h = f2bf_rne(a);
    size_t dst = (i & ~(size_t)31) + a_swz((int)(i & 31));
    xhi[dst] = h;
    xlo[dst] = f2bf_rne(a - bf2f(h));
  }
}

// Prep kernel B: W split to bf16 hi/lo, natural [col][k] layout (already
// fragment-linear for the B operand: 16 contiguous K per half-wave).
__global__ void convert_w(const float* __restrict__ Wmat,
                          unsigned short* __restrict__ whi,
                          unsigned short* __restrict__ wlo)
{
  size_t i = (size_t)blockIdx.x * blockDim.x + threadIdx.x;
  size_t stride = (size_t)gridDim.x * blockDim.x;
  const size_t n = (size_t)kH * kD;
  for (; i < n; i += stride) {
    float w = Wmat[i];
    unsigned short h = f2bf_rne(w);
    whi[i] = h;
    wlo[i] = f2bf_rne(w - bf2f(h));
  }
}

// ---------------------------------------------------------------------------
// Kernel 1: encoder GEMM  lat_pre = (x-pre_bias) @ W^T + latent_bias
// bf16 hi/lo split -> 3x v_wmma_f32_16x16x32_bf16 per (tile, K-step).
// Block = 256 threads (8 waves); wave tile = 32(M) x 64(N); block = 256(M) x 64(N).
// Staging is a pure uint4 copy of preconverted bf16 (coalesced, half the bytes);
// inner loop = ds_load_b128 fragments + v_wmma only.
// ---------------------------------------------------------------------------
__global__ void __launch_bounds__(256)
encoder_gemm_wmma(const unsigned short* __restrict__ xhi,
                  const unsigned short* __restrict__ xlo,
                  const unsigned short* __restrict__ whi,
                  const unsigned short* __restrict__ wlo,
                  const float* __restrict__ latent_bias,
                  float* __restrict__ lat_pre)
{
  alignas(16) __shared__ unsigned short sA_hi[256 * 32];  // 16 KB
  alignas(16) __shared__ unsigned short sA_lo[256 * 32];  // 16 KB
  alignas(16) __shared__ unsigned short sB_hi[64 * 32];   //  4 KB
  alignas(16) __shared__ unsigned short sB_lo[64 * 32];   //  4 KB

  const int tid  = threadIdx.x;
  const int wave = tid >> 5;
  const int lane = tid & 31;
  const int half = lane >> 4;   // half-wave (A/B/C layout split)
  const int l16  = lane & 15;

  const int nb = blockIdx.x * 64;     // column base (hidden unit)
  const int mb = blockIdx.y * 256;    // row base for this block

  // accumulators: 2 M-tiles x 4 N-tiles of 16x16 f32 (v8f each)
  v8f c[2][4];
  for (int nt = 0; nt < 4; ++nt) {
    float lb = latent_bias[nb + nt * 16 + l16];
    for (int mt = 0; mt < 2; ++mt)
      for (int e = 0; e < 8; ++e) c[mt][nt][e] = lb;
  }

  for (int kb = 0; kb < kD; kb += 32) {
    // ---- stage A tile: 256 rows x 32 k = 1024 uint4 per array (4 per thread)
    for (int i = tid; i < 1024; i += 256) {
      int r     = i >> 2;          // row in tile
      int chunk = i & 3;           // which 16B of the row's 64B k-group
      size_t src = (size_t)(mb + r) * kD + kb;   // shorts (64B aligned)
      ((uint4*)sA_hi)[i] = ((const uint4*)(xhi + src))[chunk];
      ((uint4*)sA_lo)[i] = ((const uint4*)(xlo + src))[chunk];
      if (chunk == 0) __builtin_prefetch(xhi + src + 32);  // next K-tile
    }
    // ---- stage B tile: 64 cols x 32 k = 256 uint4 per array (1 per thread)
    {
      int col   = tid >> 2;
      int chunk = tid & 3;
      size_t src = (size_t)(nb + col) * kD + kb;
      ((uint4*)sB_hi)[tid] = ((const uint4*)(whi + src))[chunk];
      ((uint4*)sB_lo)[tid] = ((const uint4*)(wlo + src))[chunk];
    }
    __syncthreads();

    // ---- A fragments: one contiguous 32B LDS read per (tile, hi/lo)
    v16bf ahi[2], alo[2];
    for (int mt = 0; mt < 2; ++mt) {
      int off = (wave * 32 + mt * 16 + l16) * 32 + half * 16;
      ahi[mt] = __builtin_bit_cast(v16bf, *(const v16us*)&sA_hi[off]);
      alo[mt] = __builtin_bit_cast(v16bf, *(const v16us*)&sA_lo[off]);
    }

    // ---- B fragments + 3-term split product: hi*hi + hi*lo + lo*hi
    for (int nt = 0; nt < 4; ++nt) {
      int off = (nt * 16 + l16) * 32 + half * 16;
      v16bf bhi = __builtin_bit_cast(v16bf, *(const v16us*)&sB_hi[off]);
      v16bf blo = __builtin_bit_cast(v16bf, *(const v16us*)&sB_lo[off]);
      for (int mt = 0; mt < 2; ++mt) {
        c[mt][nt] = __builtin_amdgcn_wmma_f32_16x16x32_bf16(
            false, ahi[mt], false, bhi, (short)0, c[mt][nt], false, false);
        c[mt][nt] = __builtin_amdgcn_wmma_f32_16x16x32_bf16(
            false, ahi[mt], false, blo, (short)0, c[mt][nt], false, false);
        c[mt][nt] = __builtin_amdgcn_wmma_f32_16x16x32_bf16(
            false, alo[mt], false, bhi, (short)0, c[mt][nt], false, false);
      }
    }
    __syncthreads();
  }

  // ---- store C: lane -> col = nb+nt*16+l16; VGPR r -> row mb+wave*32+mt*16+half*8+r
  for (int mt = 0; mt < 2; ++mt)
    for (int nt = 0; nt < 4; ++nt) {
      int col = nb + nt * 16 + l16;
      for (int r = 0; r < 8; ++r) {
        int m = mb + wave * 32 + mt * 16 + half * 8 + r;
        lat_pre[(size_t)m * kH + col] = c[mt][nt][r];
      }
    }
}

// ---------------------------------------------------------------------------
// Kernel 2: per-row top-K via 4-pass radix select on monotonic uint keys.
// One block per row. Scatters relu(val) into `scat` (pre-zeroed) and counts
// firing units (val > 1e-5) into `fired` when requested.
// ---------------------------------------------------------------------------
template <int KSEL, bool DO_FIRED>
__global__ void __launch_bounds__(256)
topk_select(const float* __restrict__ src, float* __restrict__ scat,
            int* __restrict__ fired)
{
  __shared__ unsigned int keys[kH];
  __shared__ unsigned int hist[256];
  __shared__ unsigned int sh[2];
  __shared__ unsigned int eq_taken;

  const int tid = threadIdx.x;
  const size_t base = (size_t)blockIdx.x * kH;

  for (int i = tid; i < kH; i += 256) {
    unsigned int u = __float_as_uint(src[base + i]);
    keys[i] = (u & 0x80000000u) ? ~u : (u | 0x80000000u);  // order-preserving
  }
  __syncthreads();

  unsigned int prefix = 0, want = (unsigned int)KSEL;
  for (int p = 3; p >= 0; --p) {
    const int shift = p * 8;
    hist[tid] = 0;
    __syncthreads();
    const unsigned int hmask = (p == 3) ? 0u : (0xFFFFFFFFu << (shift + 8));
    for (int i = tid; i < kH; i += 256) {
      unsigned int kk = keys[i];
      if ((kk & hmask) == prefix)
        atomicAdd(&hist[(kk >> shift) & 255u], 1u);
    }
    __syncthreads();
    if (tid == 0) {
      unsigned int acc = 0;
      int d = 255;
      for (; d > 0; --d) {
        if (acc + hist[d] >= want) break;
        acc += hist[d];
      }
      sh[0] = prefix | ((unsigned int)d << shift);
      sh[1] = want - acc;
      eq_taken = 0;
    }
    __syncthreads();
    prefix = sh[0];
    want   = sh[1];
    __syncthreads();
  }

  const unsigned int T = prefix;  // key of the KSEL-th largest
  const unsigned int r = want;    // how many ties at T to keep
  for (int i = tid; i < kH; i += 256) {
    unsigned int kk = keys[i];
    bool sel = (kk > T);
    if (!sel && kk == T) sel = (atomicAdd(&eq_taken, 1u) < r);
    if (sel) {
      float v = src[base + i];
      scat[base + i] = v > 0.f ? v : 0.f;          // relu(scatter)
      if (DO_FIRED && v > 1e-5f) atomicAdd(&fired[i], 1);
    }
  }
}

// ---------------------------------------------------------------------------
// Kernel 3: stats update  s' = fired ? 1 : s+1  (optionally export as float)
// ---------------------------------------------------------------------------
__global__ void stats_update(const int* __restrict__ s_in,
                             const int* __restrict__ fired,
                             int* __restrict__ s_out,
                             float* __restrict__ s_out_f)
{
  int h = blockIdx.x * blockDim.x + threadIdx.x;
  if (h >= kH) return;
  int s = (fired[h] > 0) ? 1 : (s_in[h] + 1);
  s_out[h] = s;
  if (s_out_f) s_out_f[h] = (float)s;
}

// ---------------------------------------------------------------------------
// Kernel 4: dead-unit mask applied in place:  lat[m,h] *= (stats[h] > 30)
// ---------------------------------------------------------------------------
__global__ void dead_mask(float* __restrict__ lat, const int* __restrict__ stats,
                          size_t n)
{
  size_t i = (size_t)blockIdx.x * blockDim.x + threadIdx.x;
  size_t stride = (size_t)gridDim.x * blockDim.x;
  for (; i < n; i += stride)
    if (stats[i & (size_t)(kH - 1)] <= kDead) lat[i] = 0.f;
}

// ---------------------------------------------------------------------------
// Kernel 5: sparse tied decode  out[m,:] = pre_bias + sum_nz val * W[idx,:]
// One block per row; compact nonzeros to LDS; W rows are L2-resident (16.8MB).
// ---------------------------------------------------------------------------
__global__ void __launch_bounds__(256)
sparse_decode(const float* __restrict__ lat, const float* __restrict__ Wmat,
              const float* __restrict__ pre_bias, float* __restrict__ out)
{
  __shared__ int   s_idx[kAux];
  __shared__ float s_val[kAux];
  __shared__ unsigned int s_cnt;

  const int tid = threadIdx.x;
  const size_t row = blockIdx.x;
  if (tid == 0) s_cnt = 0;
  __syncthreads();

  for (int i = tid; i < kH; i += 256) {
    float v = lat[row * kH + i];
    if (v != 0.f) {
      unsigned int p = atomicAdd(&s_cnt, 1u);
      if (p < (unsigned int)kAux) { s_idx[p] = i; s_val[p] = v; }
    }
  }
  __syncthreads();
  unsigned int n = s_cnt;
  if (n > (unsigned int)kAux) n = kAux;

  float acc[4];
  for (int j = 0; j < 4; ++j) acc[j] = pre_bias[tid + 256 * j];
  for (unsigned int t = 0; t < n; ++t) {
    float v = s_val[t];
    const float* wr = Wmat + (size_t)s_idx[t] * kD;
    for (int j = 0; j < 4; ++j) acc[j] += v * wr[tid + 256 * j];
  }
  for (int j = 0; j < 4; ++j) out[row * kD + tid + 256 * j] = acc[j];
}

// ---------------------------------------------------------------------------
// small utility zero kernels
// ---------------------------------------------------------------------------
__global__ void zero_f(float* __restrict__ p, size_t n) {
  size_t i = (size_t)blockIdx.x * blockDim.x + threadIdx.x;
  size_t stride = (size_t)gridDim.x * blockDim.x;
  for (; i < n; i += stride) p[i] = 0.f;
}
__global__ void zero_i(int* __restrict__ p, int n) {
  int i = blockIdx.x * blockDim.x + threadIdx.x;
  if (i < n) p[i] = 0;
}

// ---------------------------------------------------------------------------
// Host-side orchestration (graph-capture safe: launches only)
// ---------------------------------------------------------------------------
extern "C" void kernel_launch(void* const* d_in, const int* in_sizes, int n_in,
                              void* d_out, int out_size, void* d_ws, size_t ws_size,
                              hipStream_t stream) {
  (void)in_sizes; (void)n_in; (void)out_size; (void)ws_size;

  const float* x           = (const float*)d_in[0];
  const float* Wm          = (const float*)d_in[1];
  const float* pre_bias    = (const float*)d_in[2];
  const float* latent_bias = (const float*)d_in[3];
  const int*   stats_in    = (const int*)d_in[4];

  const size_t BH = (size_t)kB * kH;   // 33,554,432
  const size_t BD = (size_t)kB * kD;   //  8,388,608
  const size_t HD = (size_t)kH * kD;   //  4,194,304

  float* out     = (float*)d_out;
  float* lat_k   = out;                // [B,H]
  float* lat_4k  = out + BH;           // [B,H]
  float* lat_aux = out + 2 * BH;       // [B,H]
  float* lat_pre = out + 3 * BH;       // [B,H] (masked pre-acts, computed in place)
  float* rec_k   = out + 4 * BH;       // [B,D]
  float* rec_4k  = rec_k + BD;         // [B,D]
  float* rec_aux = rec_4k + BD;        // [B,D]
  float* stats_f = rec_aux + BD;       // [H] (int stats exported as float)

  // workspace: preconverted bf16 hi/lo operands + firing counters
  unsigned short* xhi = (unsigned short*)d_ws;   // [B*D]
  unsigned short* xlo = xhi + BD;                // [B*D]
  unsigned short* whi = xlo + BD;                // [H*D]
  unsigned short* wlo = whi + HD;                // [H*D]
  int* fired1 = (int*)(wlo + HD);                // [H]
  int* fired2 = fired1 + kH;                     // [H]
  int* stats1 = fired2 + kH;                     // [H]
  int* stats2 = stats1 + kH;                     // [H]

  // 0) zero the three scatter targets + firing counters
  zero_f<<<4096, 256, 0, stream>>>(lat_k, 3 * BH);
  zero_i<<<(2 * kH + 255) / 256, 256, 0, stream>>>(fired1, 2 * kH);

  // 1) one-shot operand conversion (x - pre_bias and W, bf16 hi/lo split)
  convert_x<<<8192, 256, 0, stream>>>(x, pre_bias, xhi, xlo);
  convert_w<<<4096, 256, 0, stream>>>(Wm, whi, wlo);

  // 2) encoder GEMM (bf16 split WMMA) -> lat_pre
  dim3 ggrid(kH / 64, kB / 256);
  encoder_gemm_wmma<<<ggrid, 256, 0, stream>>>(xhi, xlo, whi, wlo,
                                               latent_bias, lat_pre);

  // 3) top-8 -> lat_k, fired1
  topk_select<kTopK, true><<<kB, 256, 0, stream>>>(lat_pre, lat_k, fired1);

  // 4) stats pass 1
  stats_update<<<kH / 256, 256, 0, stream>>>(stats_in, fired1, stats1, nullptr);

  // 5) dead-mask 1 (in place)
  dead_mask<<<8192, 256, 0, stream>>>(lat_pre, stats1, BH);

  // 6) top-512 on masked -> lat_aux ; top-32 on masked -> lat_4k, fired2
  topk_select<kAux, false><<<kB, 256, 0, stream>>>(lat_pre, lat_aux, nullptr);
  topk_select<4 * kTopK, true><<<kB, 256, 0, stream>>>(lat_pre, lat_4k, fired2);

  // 7) stats pass 2 (+ export final stats as float)
  stats_update<<<kH / 256, 256, 0, stream>>>(stats1, fired2, stats2, stats_f);

  // 8) dead-mask 2 -> final latents_pre output
  dead_mask<<<8192, 256, 0, stream>>>(lat_pre, stats2, BH);

  // 9) sparse tied decodes (W fits in 192MB L2)
  sparse_decode<<<kB, 256, 0, stream>>>(lat_k,   Wm, pre_bias, rec_k);
  sparse_decode<<<kB, 256, 0, stream>>>(lat_4k,  Wm, pre_bias, rec_4k);
  sparse_decode<<<kB, 256, 0, stream>>>(lat_aux, Wm, pre_bias, rec_aux);
}
